// DeepsetsHead_22462678958288
// MI455X (gfx1250) — matrix operand order repaired and checked by reading, not compiled, not run on hardware
//
#include <hip/hip_runtime.h>
#include <cstdint>
#include <cstddef>

// ---------------------------------------------------------------------------
// Types for CDNA5 WMMA (wave32, 16x16x32 bf16 -> f32)
// ---------------------------------------------------------------------------
typedef __bf16 bf16;
typedef __attribute__((ext_vector_type(16))) __bf16 v16bf;
typedef __attribute__((ext_vector_type(8)))  float  v8f;
typedef uint32_t u32x4 __attribute__((ext_vector_type(4)));
typedef uint32_t u32x8 __attribute__((ext_vector_type(8)));

union FragU { v16bf v; uint4 q[2]; };

#define N_PROP   20000
#define M_REAL   1000
#define M_PAD    1024
#define FEAT     1024
#define IMG_WF   1333.0f
#define IMG_HF   800.0f
#define EPS_F    2.220446049250313e-16f

// LDS tile row stride: 32 data bf16 + 8 pad bf16 = 80B (20 banks, conflict-light).
// Matches TDM pad: pad_interval code 3 (16 DWORDs = 64B), pad_amount code 3 (4 DWORDs = 16B).
#define AS_STR 40
#define BS_STR 40

// ---------------------------------------------------------------------------
// 1) Pack weights: Wcat = [Wg | -Wl] (bf16, zero padded), bias padded (f32)
// ---------------------------------------------------------------------------
__global__ void pack_w(const float* __restrict__ Wg, const float* __restrict__ Wl,
                       const float* __restrict__ bg,
                       bf16* __restrict__ Wcat, float* __restrict__ biasp,
                       int Nreal, int Kreal, int Npad, int Kpad)
{
    const int total = Npad * 2 * Kpad;
    const int e = blockIdx.x * 256 + threadIdx.x;
    if (e >= total) return;
    const int n  = e / (2 * Kpad);
    const int k2 = e - n * (2 * Kpad);
    float v = 0.0f;
    if (k2 < Kpad) {
        if (n < Nreal && k2 < Kreal) v = Wg[(size_t)n * Kreal + k2];
    } else {
        const int k = k2 - Kpad;
        if (n < Nreal && k < Kreal) v = -Wl[(size_t)n * Kreal + k];
    }
    Wcat[(size_t)n * 2 * Kpad + k2] = (bf16)v;
    if (k2 == 0) biasp[n] = (n < Nreal) ? bg[n] : 0.0f;
}

// ---------------------------------------------------------------------------
// 2) Per-class top-1000 via full deterministic bitonic sort of 32768 (key,idx)
//    pairs in 256KB of dynamic LDS (CDNA5 WGP has 320KB).
//    Ties broken by ascending index (stable, matches lax.top_k).
// ---------------------------------------------------------------------------
__global__ void topk_kernel(const float* __restrict__ cls,
                            float* __restrict__ tkS, int* __restrict__ tkI)
{
    extern __shared__ unsigned sm[];
    unsigned* keys = sm;
    unsigned* idxs = sm + 32768;
    const int c = blockIdx.x;          // class 0..39 -> score column c+1
    const int tid = threadIdx.x;

    for (int i = tid; i < 32768; i += blockDim.x) {
        if (i < N_PROP) {
            float s = cls[(size_t)i * 41 + (c + 1)];
            keys[i] = __float_as_uint(s);   // positive floats order like uints
            idxs[i] = (unsigned)i;
        } else { keys[i] = 0u; idxs[i] = 0x7FFFFFFFu; }
    }
    for (unsigned k = 2; k <= 32768u; k <<= 1) {
        for (unsigned j = k >> 1; j > 0; j >>= 1) {
            __syncthreads();
            for (unsigned t = tid; t < 16384u; t += blockDim.x) {
                unsigned i  = ((t & ~(j - 1u)) << 1) | (t & (j - 1u));
                unsigned ix = i | j;
                unsigned ka = keys[i], kb = keys[ix];
                unsigned ia = idxs[i], ib = idxs[ix];
                bool lessA = (ka < kb) || (ka == kb && ia > ib); // composite (key, -idx)
                bool desc  = ((i & k) == 0u);
                if (desc ? lessA : !lessA) {
                    keys[i] = kb; keys[ix] = ka;
                    idxs[i] = ib; idxs[ix] = ia;
                }
            }
        }
    }
    __syncthreads();
    for (int r = tid; r < M_PAD; r += blockDim.x) {
        if (r < M_REAL) {
            tkS[c * M_PAD + r] = __uint_as_float(keys[r]);
            tkI[c * M_PAD + r] = (int)idxs[r];
        } else { tkS[c * M_PAD + r] = 0.0f; tkI[c * M_PAD + r] = 0; }
    }
}

// ---------------------------------------------------------------------------
// 3) Gather + build input features [1024 x 1056] bf16 (row-major + transpose)
//    cols: 0-3 bb, 4-1027 feats, 1028 w, 1029 h, 1030 ar, 1031 area, 1032 sc
// ---------------------------------------------------------------------------
__global__ void gather_build(const float* __restrict__ mb,
                             const float* __restrict__ tkS,
                             const int*   __restrict__ tkI,
                             const float* __restrict__ feats,
                             bf16* __restrict__ xc1, long long sXC,
                             bf16* __restrict__ xt1, long long sXT)
{
    const int c   = blockIdx.y;
    const int tid = threadIdx.x;
    bf16* xc = xc1 + (size_t)c * sXC;
    bf16* xt = xt1 + (size_t)c * sXT;
    const int r0 = blockIdx.x * 128;
    for (int r = r0; r < r0 + 128; ++r) {
        bf16* xr = xc + (size_t)r * 2112;
        if (r < M_REAL) {
            const int id = tkI[c * M_PAD + r];
            const float b0 = mb[(size_t)id * 164 + (c + 1) * 4 + 0];
            const float b1 = mb[(size_t)id * 164 + (c + 1) * 4 + 1];
            const float b2 = mb[(size_t)id * 164 + (c + 1) * 4 + 2];
            const float b3 = mb[(size_t)id * 164 + (c + 1) * 4 + 3];
            const float sc = tkS[c * M_PAD + r];
            const float x1 = b0 / IMG_WF, y1 = b1 / IMG_HF;
            const float x2 = b2 / IMG_WF, y2 = b3 / IMG_HF;
            const float w  = (x2 - x1) / IMG_WF;
            const float h  = (y2 - y1) / IMG_HF;
            const float ar = w / (h + EPS_F);
            const float area = w * h;
            for (int k = tid; k < 1056; k += 256) {
                float v;
                if      (k < 4)    v = (k == 0) ? b0 : (k == 1) ? b1 : (k == 2) ? b2 : b3;
                else if (k < 1028) v = feats[(size_t)id * FEAT + (k - 4)];
                else if (k == 1028) v = w;
                else if (k == 1029) v = h;
                else if (k == 1030) v = ar;
                else if (k == 1031) v = area;
                else if (k == 1032) v = sc;
                else v = 0.0f;
                bf16 hv = (bf16)v;
                xr[k] = hv;
                xt[(size_t)k * M_PAD + r] = hv;
            }
        } else {
            for (int k = tid; k < 1056; k += 256) {
                xr[k] = (bf16)0.0f;
                xt[(size_t)k * M_PAD + r] = (bf16)0.0f;
            }
        }
    }
}

// ---------------------------------------------------------------------------
// 4) Greedy sequential IoU clustering — one wave32 per class.
//    Single-wave workgroup: barrier ops lower to S_NOP (ISA 3.1).
// ---------------------------------------------------------------------------
__global__ void cluster_kernel(const float* __restrict__ mb,
                               const int* __restrict__ tkI,
                               int* __restrict__ cidG, int* __restrict__ cntG)
{
    const int c   = blockIdx.x;
    const int tid = threadIdx.x;    // 0..31
    __shared__ float X1[M_REAL], Y1[M_REAL], X2[M_REAL], Y2[M_REAL], AREA_S[M_REAL];
    __shared__ int freeF[M_REAL];
    __shared__ int cidS[M_PAD];
    __shared__ int cnt[M_PAD];
    __shared__ int curc;

    for (int r = tid; r < M_PAD; r += 32) { cidS[r] = 0; cnt[r] = 0; }
    for (int r = tid; r < M_REAL; r += 32) {
        const int id = tkI[c * M_PAD + r];
        const float b0 = mb[(size_t)id * 164 + (c + 1) * 4 + 0];
        const float b1 = mb[(size_t)id * 164 + (c + 1) * 4 + 1];
        const float b2 = mb[(size_t)id * 164 + (c + 1) * 4 + 2];
        const float b3 = mb[(size_t)id * 164 + (c + 1) * 4 + 3];
        X1[r] = b0; Y1[r] = b1; X2[r] = b2; Y2[r] = b3;
        AREA_S[r] = (b2 - b0) * (b3 - b1);
        freeF[r] = 1;
    }
    if (tid == 0) curc = 0;
    __syncthreads();

    for (int j = 0; j < M_REAL; ++j) {
        __syncthreads();
        if (!freeF[j]) continue;                      // uniform across the wave
        const float jx1 = X1[j], jy1 = Y1[j], jx2 = X2[j], jy2 = Y2[j], ja = AREA_S[j];
        const int cc = curc;
        __syncthreads();
        for (int r = tid; r < M_REAL; r += 32) {
            if (!freeF[r]) continue;
            const float iw = fminf(jx2, X2[r]) - fmaxf(jx1, X1[r]);
            const float ih = fminf(jy2, Y2[r]) - fmaxf(jy1, Y1[r]);
            const float inter = fmaxf(iw, 0.0f) * fmaxf(ih, 0.0f);
            const float iou = inter / (ja + AREA_S[r] - inter + 1e-12f);
            if (iou > 0.5f) { cidS[r] = cc; freeF[r] = 0; }
        }
        if (tid == 0) curc = cc + 1;
    }
    __syncthreads();
    for (int r = tid; r < M_REAL; r += 32) atomicAdd(&cnt[cidS[r]], 1);
    __syncthreads();
    for (int r = tid; r < M_PAD; r += 32) {
        cidG[c * M_PAD + r] = cidS[r];
        cntG[c * M_PAD + r] = (cnt[r] > 0) ? cnt[r] : 1;
    }
}

// ---------------------------------------------------------------------------
// 5) Build cluster-averaging matrix P [1024x1024] bf16 per class
// ---------------------------------------------------------------------------
__global__ void build_p(const int* __restrict__ cidG, const int* __restrict__ cntG,
                        bf16* __restrict__ P)
{
    const int c = blockIdx.y;
    const int e = blockIdx.x * 256 + threadIdx.x;  // < 1048576
    const int i = e >> 10, j = e & 1023;
    const int ci = cidG[c * M_PAD + i];
    const int cj = cidG[c * M_PAD + j];
    float v = 0.0f;
    if (i < M_REAL && j < M_REAL && ci == cj)
        v = 1.0f / (float)cntG[c * M_PAD + ci];
    P[(size_t)c * 1048576 + e] = (bf16)v;
}

// ---------------------------------------------------------------------------
// 6) bf16 WMMA GEMM: C[1024 x N] = A[1024 x K] @ B[N x K]^T (+bias, ELU)
//    Block tile 256x32, 8 waves, each wave 32x32 (2x2 x WMMA 16x16x32 bf16).
//    Double-buffered LDS, A tile fed by the Tensor Data Mover (TDM DMA with
//    LDS padding), B tile fed by global_load_async_to_lds_b128 (ASYNCcnt).
//    N, K multiples of 32; M = 1024 -> no guards, EXEC all-ones for WMMA.
// ---------------------------------------------------------------------------
__global__ __launch_bounds__(256) void gemm_bf16(
    const bf16* __restrict__ A, long long sA,
    const bf16* __restrict__ B, long long sB,
    bf16* __restrict__ C, long long sC,
    bf16* __restrict__ CT, long long sCT,
    const float* __restrict__ bias,
    int K, int lda, int ldb, int ldc, int ldct, int doElu)
{
    __shared__ __align__(16) bf16 As[2][256 * AS_STR];
    __shared__ __align__(16) bf16 Bs[2][32 * BS_STR];

    const int cc = blockIdx.z;
    A += (size_t)cc * sA;
    B += (size_t)cc * sB;
    C += (size_t)cc * sC;
    if (CT) CT += (size_t)cc * sCT;

    const int tid  = threadIdx.x;
    const int wave = tid >> 5;
    const int lane = tid & 31;
    const int llo  = lane & 15;
    const int lhi  = lane >> 4;
    const int m0b  = blockIdx.y * 256;
    const int n0b  = blockIdx.x * 32;
    const int ntk  = K >> 5;

    // TDM descriptor group 1 (constant across tiles):
    //   data_size=2B, pad_enable, pad_interval=64B, pad_amount=16B,
    //   tensor 32 x 256 (dim0 x dim1), tile 32 x 256, dim0_stride = lda
    u32x8 g1;
    g1[0] = (1u << 16) | (1u << 20) | (3u << 22) | (3u << 25);
    g1[1] = 32u << 16;            // tensor_dim0[15:0]
    g1[2] = 256u << 16;           // tensor_dim0[31:16]=0 | tensor_dim1[15:0]
    g1[3] = 32u << 16;            // tensor_dim1[31:16]=0 | tile_dim0
    g1[4] = 256u;                 // tile_dim1 | tile_dim2=0
    g1[5] = (uint32_t)lda;        // tensor_dim0_stride[31:0]
    g1[6] = 0u;
    g1[7] = 0u;

    auto issue_tile = [&](int kt, int buf) {
        const int k0 = kt << 5;
        if (wave == 0) {
            // A tile 256x32 via Tensor Data Mover (one DMA per block per tile)
            const uint64_t ga = (uint64_t)(uintptr_t)(A + (size_t)m0b * lda + k0);
            const uint32_t la = (uint32_t)(uintptr_t)&As[buf][0];
            u32x4 g0;
            g0[0] = 1u;                                   // count=1, user descriptor
            g0[1] = la;                                   // lds_addr
            g0[2] = (uint32_t)ga;                         // global_addr[31:0]
            g0[3] = ((uint32_t)(ga >> 32) & 0x01FFFFFFu)  // global_addr[56:32]
                    | 0x80000000u;                        // type=2 ("image")
            asm volatile("tensor_load_to_lds %0, %1" :: "s"(g0), "s"(g1) : "memory");
        }
        // B tile 32x32: per-lane async DMA, no VGPR round-trip
        if (tid < 128) {
            const int row = tid >> 2, colc = (tid & 3) << 3;
            const uint64_t gb =
                (uint64_t)(uintptr_t)(B + (size_t)(n0b + row) * ldb + k0 + colc);
            const uint32_t lb = (uint32_t)(uintptr_t)&Bs[buf][row * BS_STR + colc];
            asm volatile("global_load_async_to_lds_b128 %0, %1, off"
                         :: "v"(lb), "v"(gb) : "memory");
        }
    };

    v8f acc[2][2];
#pragma unroll
    for (int mi = 0; mi < 2; ++mi)
#pragma unroll
        for (int ni = 0; ni < 2; ++ni)
#pragma unroll
            for (int e = 0; e < 8; ++e) acc[mi][ni][e] = 0.0f;

    issue_tile(0, 0);
    for (int kt = 0; kt < ntk; ++kt) {
        const int buf = kt & 1;
        // own async B-chunks done; own TDM (wave 0) done; barrier publishes all
        asm volatile("s_wait_asynccnt 0x0" ::: "memory");
        if (wave == 0) __builtin_amdgcn_s_wait_tensorcnt(0);
        __syncthreads();
        if (kt + 1 < ntk) issue_tile(kt + 1, buf ^ 1);

        // A fragment: lane holds row m=llo(+16*mi), K = 8*lhi + e (+16 for e>=8)
        FragU fa[2], fb[2];
#pragma unroll
        for (int mi = 0; mi < 2; ++mi) {
            const bf16* ap = &As[buf][(wave * 32 + mi * 16 + llo) * AS_STR + lhi * 8];
            fa[mi].q[0] = *(const uint4*)ap;
            fa[mi].q[1] = *(const uint4*)(ap + 16);
        }
        // B fragment: lane holds col n=llo(+16*ni), K = 8*lhi + e (+16 for e>=8)
#pragma unroll
        for (int ni = 0; ni < 2; ++ni) {
            const bf16* bp = &Bs[buf][(ni * 16 + llo) * BS_STR + lhi * 8];
            fb[ni].q[0] = *(const uint4*)bp;
            fb[ni].q[1] = *(const uint4*)(bp + 16);
        }
#pragma unroll
        for (int mi = 0; mi < 2; ++mi)
#pragma unroll
            for (int ni = 0; ni < 2; ++ni)
                acc[mi][ni] = __builtin_amdgcn_wmma_f32_16x16x32_bf16(
                    false, fa[mi].v, false, fb[ni].v, (short)0, acc[mi][ni],
                    false, false);
    }

    // Epilogue: C/D layout — VGPR v holds M = v + 8*lhi, N = llo
#pragma unroll
    for (int mi = 0; mi < 2; ++mi) {
#pragma unroll
        for (int ni = 0; ni < 2; ++ni) {
            const int col = n0b + ni * 16 + llo;
            const float bz = bias ? bias[col] : 0.0f;
            const int mbase = m0b + wave * 32 + mi * 16 + lhi * 8;
#pragma unroll
            for (int e = 0; e < 8; ++e) {
                float x = acc[mi][ni][e] + bz;
                if (doElu) x = (x > 0.0f) ? x : (__expf(x) - 1.0f);
                const bf16 h = (bf16)x;
                const int m = mbase + e;
                C[(size_t)m * ldc + col] = h;
                if (CT) CT[(size_t)col * ldct + m] = h;
            }
        }
    }
}

// ---------------------------------------------------------------------------
// 7) Finalize: pred col 0, rows 0..999 -> d_out f32 [40,1000,1]
// ---------------------------------------------------------------------------
__global__ void finalize(const bf16* __restrict__ prd, long long sPrd,
                         float* __restrict__ out)
{
    const int c = blockIdx.x;
    for (int r = threadIdx.x; r < M_REAL; r += 256)
        out[c * M_REAL + r] = (float)prd[(size_t)c * sPrd + (size_t)r * 32];
}

// ---------------------------------------------------------------------------
// Host launcher
// ---------------------------------------------------------------------------
extern "C" void kernel_launch(void* const* d_in, const int* in_sizes, int n_in,
                              void* d_out, int out_size, void* d_ws, size_t ws_size,
                              hipStream_t stream)
{
    (void)in_sizes; (void)n_in; (void)out_size; (void)ws_size;
    const float* mb  = (const float*)d_in[0];
    const float* cls = (const float*)d_in[1];
    const float* ft  = (const float*)d_in[2];
    const float* Wg1 = (const float*)d_in[3];
    const float* bg1 = (const float*)d_in[4];
    const float* Wl1 = (const float*)d_in[5];
    const float* Wg2 = (const float*)d_in[6];
    const float* bg2 = (const float*)d_in[7];
    const float* Wl2 = (const float*)d_in[8];
    const float* Wg3 = (const float*)d_in[9];
    const float* bg3 = (const float*)d_in[10];
    const float* Wl3 = (const float*)d_in[11];
    const float* Wg4 = (const float*)d_in[12];
    const float* bg4 = (const float*)d_in[13];
    const float* Wl4 = (const float*)d_in[14];
    float* out = (float*)d_out;

    char* ws = (char*)d_ws;
    size_t off = 0;
    auto carve = [&](size_t bytes) -> char* {
        char* p = ws + off;
        off += (bytes + 255) & ~(size_t)255;
        return p;
    };

    float* tkS = (float*)carve((size_t)40 * 1024 * 4);
    int*   tkI = (int*)  carve((size_t)40 * 1024 * 4);
    int*   cid = (int*)  carve((size_t)40 * 1024 * 4);
    int*   cnt = (int*)  carve((size_t)40 * 1024 * 4);
    bf16*  P   = (bf16*) carve((size_t)40 * 1024 * 1024 * 2);

    const long long sXC1 = 1024LL * 2112;  // xcat1: [x1 | u1], also hosts xc3/xc4/pred
    const long long sXT1 = 1056LL * 1024;  // x1^T, also hosts xt3/xt4
    const long long sXC2 = 1024LL * 2048;  // xcat2: [h1 | u2]
    const long long sXT2 = 1024LL * 1024;  // h1^T
    bf16* xc1 = (bf16*)carve((size_t)40 * sXC1 * 2);
    bf16* xt1 = (bf16*)carve((size_t)40 * sXT1 * 2);
    bf16* xc2 = (bf16*)carve((size_t)40 * sXC2 * 2);
    bf16* xt2 = (bf16*)carve((size_t)40 * sXT2 * 2);

    bf16* Wc1 = (bf16*)carve((size_t)1024 * 2112 * 2);
    bf16* Wc2 = (bf16*)carve((size_t)608 * 2048 * 2);
    bf16* Wc3 = (bf16*)carve((size_t)320 * 1216 * 2);
    bf16* Wc4 = (bf16*)carve((size_t)32 * 640 * 2);
    float* bp1 = (float*)carve(1024 * 4);
    float* bp2 = (float*)carve(608 * 4);
    float* bp3 = (float*)carve(320 * 4);
    float* bp4 = (float*)carve(32 * 4);

    // Aliased sub-buffers (lifetimes are disjoint; stream order serializes):
    bf16* xc3 = xc1;                                        // [1024x1216] stride sXC1
    bf16* xc4 = xc1 + (size_t)1024 * 1216;                  // [1024x640]
    bf16* prd = xc4 + (size_t)1024 * 640;                   // [1024x32]
    bf16* xt3 = xt1;                                        // [608x1024]  stride sXT1
    bf16* xt4 = xt1 + (size_t)608 * 1024;                   // [320x1024]

    // 1) weight packing
    pack_w<<<dim3((1024 * 2112 + 255) / 256), 256, 0, stream>>>(Wg1, Wl1, bg1, Wc1, bp1, 1000, 1033, 1024, 1056);
    pack_w<<<dim3((608 * 2048 + 255) / 256), 256, 0, stream>>>(Wg2, Wl2, bg2, Wc2, bp2, 600, 1000, 608, 1024);
    pack_w<<<dim3((320 * 1216 + 255) / 256), 256, 0, stream>>>(Wg3, Wl3, bg3, Wc3, bp3, 300, 600, 320, 608);
    pack_w<<<dim3((32 * 640 + 255) / 256), 256, 0, stream>>>(Wg4, Wl4, bg4, Wc4, bp4, 1, 300, 32, 320);

    // 2) per-class sorted top-1000 (256KB dynamic LDS; CDNA5 WGP has 320KB)
    hipFuncSetAttribute((const void*)topk_kernel,
                        hipFuncAttributeMaxDynamicSharedMemorySize, 262144);
    topk_kernel<<<40, 1024, 262144, stream>>>(cls, tkS, tkI);

    // 3) features, 4) clustering, 5) P matrix
    gather_build<<<dim3(8, 40), 256, 0, stream>>>(mb, tkS, tkI, ft, xc1, sXC1, xt1, sXT1);
    cluster_kernel<<<40, 32, 0, stream>>>(mb, tkI, cid, cnt);
    build_p<<<dim3(4096, 40), 256, 0, stream>>>(cid, cnt, P);

    const long long sP = 1024LL * 1024;
    // layer 1: u1 = P @ x1 ; z1 = [x1|u1] @ Wc1^T + b1, ELU
    gemm_bf16<<<dim3(33, 4, 40), 256, 0, stream>>>(P, sP, xt1, sXT1,
        xc1 + 1056, sXC1, nullptr, 0, nullptr, 1024, 1024, 1024, 2112, 0, 0);
    gemm_bf16<<<dim3(32, 4, 40), 256, 0, stream>>>(xc1, sXC1, Wc1, 0,
        xc2, sXC2, xt2, sXT2, bp1, 2112, 2112, 2112, 2048, 1024, 1);
    // layer 2
    gemm_bf16<<<dim3(32, 4, 40), 256, 0, stream>>>(P, sP, xt2, sXT2,
        xc2 + 1024, sXC2, nullptr, 0, nullptr, 1024, 1024, 1024, 2048, 0, 0);
    gemm_bf16<<<dim3(19, 4, 40), 256, 0, stream>>>(xc2, sXC2, Wc2, 0,
        xc3, sXC1, xt3, sXT1, bp2, 2048, 2048, 2048, 1216, 1024, 1);
    // layer 3
    gemm_bf16<<<dim3(19, 4, 40), 256, 0, stream>>>(P, sP, xt3, sXT1,
        xc3 + 608, sXC1, nullptr, 0, nullptr, 1024, 1024, 1024, 1216, 0, 0);
    gemm_bf16<<<dim3(10, 4, 40), 256, 0, stream>>>(xc3, sXC1, Wc3, 0,
        xc4, sXC1, xt4, sXT1, bp3, 1216, 1216, 1216, 640, 1024, 1);
    // layer 4
    gemm_bf16<<<dim3(10, 4, 40), 256, 0, stream>>>(P, sP, xt4, sXT1,
        xc4 + 320, sXC1, nullptr, 0, nullptr, 1024, 1024, 1024, 640, 0, 0);
    gemm_bf16<<<dim3(1, 4, 40), 256, 0, stream>>>(xc4, sXC1, Wc4, 0,
        prd, sXC1, nullptr, 0, bp4, 640, 640, 640, 32, 0, 1);

    finalize<<<40, 256, 0, stream>>>(prd, sXC1, out);
}
